// encoder_rnn_59811714564494
// MI455X (gfx1250) — matrix operand order repaired and checked by reading, not compiled, not run on hardware
//
#include <hip/hip_runtime.h>
#include <stdint.h>

// Problem constants (from the reference).
#define T_STEPS 8192
#define DIN     512
#define H       768
#define G3      2304   // 3*H
#define H2      1536   // 2*H
#define DOUT    256
#define NWG_REC 8      // workgroups per direction in the recurrence relay
#define HS      (H / NWG_REC)   // 96 h-elements per WG

typedef __attribute__((ext_vector_type(16))) __bf16 bf16x16;
typedef __attribute__((ext_vector_type(8)))  __bf16 bf16x8;
typedef __attribute__((ext_vector_type(8)))  float  f32x8;
typedef __attribute__((ext_vector_type(4)))  float  f32x4;

#if defined(__HIP_DEVICE_COMPILE__) && __has_builtin(__builtin_amdgcn_tensor_load_to_lds)
#define USE_TDM 1
#else
#define USE_TDM 0
#endif

// ---------------------------------------------------------------------------
// Utility kernels
// ---------------------------------------------------------------------------
__global__ void k_zero_ints(int* p, int n) {
    int i = blockIdx.x * blockDim.x + threadIdx.x;
    if (i < n) p[i] = 0;
}

__global__ void k_f32_to_bf16(const float* __restrict__ src,
                              __bf16* __restrict__ dst, size_t n) {
    size_t stride = (size_t)gridDim.x * blockDim.x;
    for (size_t i = (size_t)blockIdx.x * blockDim.x + threadIdx.x; i < n; i += stride)
        dst[i] = (__bf16)src[i];
}

// ---------------------------------------------------------------------------
// TDM descriptor build + issue (CDNA5 ISA ch.8: D# groups).
// Loads a 2D tile: tile_dim0 = 32 elems (K), tile_dim1 = `rows`, element = 2B,
// from a row-major [tensor_dim1 x K] bf16 tensor, into LDS at lds_off.
// Must be executed by exactly one wave (TDM ignores EXEC).
// ---------------------------------------------------------------------------
#if USE_TDM
typedef __attribute__((ext_vector_type(4))) unsigned int u32x4;
typedef __attribute__((ext_vector_type(8))) int          i32x8;
typedef __attribute__((ext_vector_type(4))) int          i32x4;
typedef __attribute__((address_space(3))) __bf16         lds_bf16_t;

__device__ __forceinline__ void tdm_load_tile_a(unsigned lds_off,
                                                const __bf16* gptr,
                                                int K, int M, int rows) {
    unsigned long long ga = (unsigned long long)(uintptr_t)gptr;
    u32x4 g0;
    g0[0] = 1u;                                   // count=1, user descriptor
    g0[1] = lds_off;                              // lds_addr  (bits 63:32)
    g0[2] = (unsigned)(ga & 0xFFFFFFFFu);         // global_addr[31:0]
    g0[3] = (unsigned)((ga >> 32) & 0x01FFFFFFu)  // global_addr[56:32]
          | (2u << 30);                           // type = 2 ("image")
    i32x8 g1;
    g1[0] = (1 << 16);                            // data_size=1 (2B), wgmask=0
    g1[1] = (K & 0xFFFF) << 16;                   // tensor_dim0[15:0] @63:48
    g1[2] = ((unsigned)K >> 16)                   // tensor_dim0[31:16] @79:64
          | ((M & 0xFFFF) << 16);                 // tensor_dim1[15:0] @95:80
    g1[3] = ((unsigned)M >> 16)                   // tensor_dim1[31:16] @111:96
          | (32 << 16);                           // tile_dim0=32 @127:112
    g1[4] = rows & 0xFFFF;                        // tile_dim1 @143:128 (dim2=0)
    g1[5] = K;                                    // tensor_dim0_stride[31:0]
    g1[6] = 0;                                    // stride[47:32], dim1_stride lo
    g1[7] = 0;
    i32x4 z4 = {0, 0, 0, 0};                      // groups 2/3 unused (2D)
    i32x8 z8 = {0, 0, 0, 0, 0, 0, 0, 0};
    __builtin_amdgcn_tensor_load_to_lds(g0, g1, z4, z4, z8, 0);
}
#endif

// Fallback cooperative stage (all 256 threads): tile[r][0..31] <- Ablk[r][k0..k0+31]
__device__ __forceinline__ void stage_tile_a(__bf16* dst, const __bf16* Ablk,
                                             int K, int k0) {
    for (int c = threadIdx.x; c < (128 * 32) / 8; c += 256) {
        const int row  = c >> 2;
        const int coff = (c & 3) * 8;
        *(bf16x8*)(dst + row * 32 + coff) =
            *(const bf16x8*)(Ablk + (size_t)row * K + k0 + coff);
    }
}

// ---------------------------------------------------------------------------
// WMMA GEMM:  C[m,n] = sum_k A[m,k] * B[n,k]  (+ bias[n])
//   A: [M x K] bf16 row-major, B: [N x K] bf16 row-major (C = A * B^T)
//   C: f32, leading dim ldc.  M%128==0, N%128==0, K%32==0 assumed.
// Block: 256 threads = 8 waves; block tile 128x128; wave tile 64(M) x 32(N).
// A tile double-buffered in LDS via the Tensor Data Mover; B direct from L2.
// ---------------------------------------------------------------------------
__global__ __launch_bounds__(256)
void k_wmma_gemm_bias(const __bf16* __restrict__ A,
                      const __bf16* __restrict__ B,
                      const float*  __restrict__ bias,
                      float* __restrict__ C, int ldc,
                      int M, int K) {
    __shared__ __align__(16) __bf16 tileA[2][128 * 32];   // 2 x 8KB

    const int lane = threadIdx.x & 31;
    const int wave = threadIdx.x >> 5;
    const int wm   = wave & 1;          // 0..1 -> 64-row groups
    const int wn   = wave >> 1;         // 0..3 -> 32-col groups
    const int m0   = blockIdx.x * 128 + wm * 64;
    const int n0   = blockIdx.y * 128 + wn * 32;
    const __bf16* Ablk = A + (size_t)(blockIdx.x * 128) * K;

    const f32x8 vzero = {0.f,0.f,0.f,0.f,0.f,0.f,0.f,0.f};
    f32x8 acc[4][2];
#pragma unroll
    for (int mt = 0; mt < 4; ++mt)
#pragma unroll
        for (int nt = 0; nt < 2; ++nt) acc[mt][nt] = vzero;

    // bf16 fragment addressing used here (contiguous k, same for A and B):
    //  A 16x32: lane m=lane&15; K = base + j, j=0..15 (contig),
    //           base = (lane>=16) ? 16 : 0.
    //  B 32x16: lane n=lane&15; K = (lane>=16 ? 16 : 0) + j, j=0..15 (contig).
    const int am    = lane & 15;
    const int abase = (lane >> 4) << 4;    // 0 or 16
    const int bn    = lane & 15;
    const int bbase = (lane >> 4) << 4;    // 0 or 16

#if USE_TDM
    if (wave == 0)
        tdm_load_tile_a((unsigned)(uintptr_t)(lds_bf16_t*)&tileA[0][0],
                        Ablk, K, M, 128);
#else
    stage_tile_a(&tileA[0][0], Ablk, K, 0);
#endif

    int cur = 0;
    for (int k0 = 0; k0 < K; k0 += 32) {
#if USE_TDM
        if (wave == 0) __builtin_amdgcn_s_wait_tensorcnt(0);
#endif
        __syncthreads();   // tileA[cur] ready; previous readers of [cur^1] done

        const int nk = k0 + 32;
        if (nk < K) {
#if USE_TDM
            if (wave == 0)
                tdm_load_tile_a((unsigned)(uintptr_t)(lds_bf16_t*)&tileA[cur ^ 1][0],
                                Ablk + nk, K, M, 128);
#else
            stage_tile_a(&tileA[cur ^ 1][0], Ablk, K, nk);
#endif
        }

        // A fragments from LDS (ds_load_b128 x2 per tile).
        bf16x16 afrag[4];
        const __bf16* tbase = &tileA[cur][0];
#pragma unroll
        for (int mt = 0; mt < 4; ++mt) {
            const int rl = wm * 64 + mt * 16 + am;
            const __bf16* tp = tbase + rl * 32 + abase;
            bf16x8 lo = *(const bf16x8*)tp;
            bf16x8 hi = *(const bf16x8*)(tp + 8);
            bf16x16 f;
#pragma unroll
            for (int i = 0; i < 8; ++i) { f[i] = lo[i]; f[i + 8] = hi[i]; }
            afrag[mt] = f;
        }
        // B fragments direct from global (weights hot in the 192MB L2).
        bf16x16 bfrag[2];
#pragma unroll
        for (int nt = 0; nt < 2; ++nt) {
            const __bf16* bp = B + (size_t)(n0 + nt * 16 + bn) * K + k0 + bbase;
            bfrag[nt] = *(const bf16x16*)bp;
        }
#pragma unroll
        for (int mt = 0; mt < 4; ++mt)
#pragma unroll
            for (int nt = 0; nt < 2; ++nt)
                acc[mt][nt] = __builtin_amdgcn_wmma_f32_16x16x32_bf16(
                    false, afrag[mt], false, bfrag[nt],
                    (short)0, acc[mt][nt], false, false);
        cur ^= 1;
    }

    // C/D layout: VGPR r -> M = r + (lane>=16 ? 8 : 0); N = lane&15.
    const int rbase = (lane < 16) ? 0 : 8;
#pragma unroll
    for (int nt = 0; nt < 2; ++nt) {
        const int col = n0 + nt * 16 + (lane & 15);
        const float bv = bias ? bias[col] : 0.f;
#pragma unroll
        for (int mt = 0; mt < 4; ++mt) {
#pragma unroll
            for (int r = 0; r < 8; ++r) {
                const int row = m0 + mt * 16 + rbase + r;
                C[(size_t)row * ldc + col] = acc[mt][nt][r] + bv;
            }
        }
    }
}

// ---------------------------------------------------------------------------
// Cross-workgroup relay barrier (agent scope, monotone phase counter).
// ---------------------------------------------------------------------------
__device__ inline void relay_barrier(int* ctr, int target) {
    __syncthreads();
    if (threadIdx.x == 0) {
        __hip_atomic_fetch_add(ctr, 1, __ATOMIC_RELEASE, __HIP_MEMORY_SCOPE_AGENT);
        while (__hip_atomic_load(ctr, __ATOMIC_ACQUIRE, __HIP_MEMORY_SCOPE_AGENT) < target)
            __builtin_amdgcn_s_sleep(2);
    }
    __syncthreads();
}

// ---------------------------------------------------------------------------
// Persistent GRU recurrence for one layer (both directions).
// Grid = 2*NWG_REC blocks x 256 threads. Each WG owns HS h-elements of one
// direction (288 gate rows of w_hh), relaying h through L2 each step.
// GI: precomputed w_ih@x + b_ih, layout [T][2][G3].
// ---------------------------------------------------------------------------
__global__ __launch_bounds__(256)
void k_gru_recurrence(const float* __restrict__ GI,
                      const float* __restrict__ w_hh,   // [2][G3][H]
                      const float* __restrict__ b_hh,   // [2][G3]
                      const float* __restrict__ h_init, // [2][H]
                      float* __restrict__ Hseq,         // store_all ? [T][2H] : [2H]
                      float* __restrict__ hbuf,         // [2][H] relay buffer
                      int*   __restrict__ sync_ctr,     // [>=2] phase counters
                      int store_all, int T) {
    __shared__ __align__(16) float h_s[H];
    __shared__ float gh_s[3 * HS];

    const int dir  = blockIdx.x / NWG_REC;
    const int wg   = blockIdx.x % NWG_REC;
    const int j0   = wg * HS;
    const int tid  = threadIdx.x;
    const int lane = tid & 31;
    const int wave = tid >> 5;

    int* ctr = sync_ctr + dir;
    const float* W   = w_hh + (size_t)dir * G3 * H;
    const float* Bh  = b_hh + (size_t)dir * G3;
    const float* GId = GI   + (size_t)dir * G3;

    // Publish this WG's slice of the initial hidden state.
    if (tid < HS) hbuf[dir * H + j0 + tid] = h_init[dir * H + j0 + tid];
    relay_barrier(ctr, NWG_REC);           // phase 1: full h0 visible

    for (int t = 0; t < T; ++t) {
        // Pull the full (768-element) hidden state into LDS.
        for (int j = tid; j < H; j += 256) h_s[j] = hbuf[dir * H + j];
        __syncthreads();

        // 288 gate-row dot products, 36 per wave, lanes split K (float4).
        const int rows_per_wave = (3 * HS) / 8;   // 36
        for (int i = 0; i < rows_per_wave; ++i) {
            const int li  = wave * rows_per_wave + i;
            const int sel = li / HS;               // 0:r 1:z 2:n
            const int j   = j0 + (li - sel * HS);
            const int row = sel * H + j;
            const float* wr = W + (size_t)row * H;
            float accum = 0.f;
#pragma unroll
            for (int c = 0; c < H / 128; ++c) {    // 6 iters: 32 lanes x float4
                f32x4 wv = *(const f32x4*)(wr + c * 128 + lane * 4);
                f32x4 hv = *(const f32x4*)(&h_s[c * 128 + lane * 4]);
                accum += wv.x * hv.x + wv.y * hv.y + wv.z * hv.z + wv.w * hv.w;
            }
#pragma unroll
            for (int off = 16; off; off >>= 1)
                accum += __shfl_xor(accum, off, 32);
            if (lane == 0) gh_s[li] = accum + Bh[row];
        }
        __syncthreads();

        // Gate combine for this WG's HS hidden elements.
        if (tid < HS) {
            const int j = j0 + tid;
            const float* gi = GId + (size_t)t * (2 * G3);
            const float i_r = gi[j];
            const float i_z = gi[H + j];
            const float i_n = gi[2 * H + j];
            const float r = 1.f / (1.f + __expf(-(i_r + gh_s[tid])));
            const float z = 1.f / (1.f + __expf(-(i_z + gh_s[HS + tid])));
            const float n = tanhf(i_n + r * gh_s[2 * HS + tid]);
            const float hp = h_s[j];
            const float hn = (1.f - z) * n + z * hp;
            hbuf[dir * H + j] = hn;
            if (store_all)          Hseq[(size_t)t * H2 + dir * H + j] = hn;
            else if (t == T - 1)    Hseq[dir * H + j] = hn;
        }
        relay_barrier(ctr, NWG_REC * (t + 2));
    }
}

// ---------------------------------------------------------------------------
// Final FC + sigmoid: y = sigmoid(fc_w @ h + fc_b), 256 x 1536.
// ---------------------------------------------------------------------------
__global__ __launch_bounds__(256)
void k_fc_sigmoid(const float* __restrict__ h,     // [H2]
                  const float* __restrict__ fc_w,  // [DOUT][H2]
                  const float* __restrict__ fc_b,  // [DOUT]
                  float* __restrict__ y) {
    __shared__ float hs[H2];
    for (int i = threadIdx.x; i < H2; i += 256) hs[i] = h[i];
    __syncthreads();
    const int o = threadIdx.x;   // DOUT == blockDim == 256
    const float* wr = fc_w + (size_t)o * H2;
    float acc = 0.f;
    for (int k = 0; k < H2; ++k) acc += wr[k] * hs[k];
    y[o] = 1.f / (1.f + __expf(-(acc + fc_b[o])));
}

// ---------------------------------------------------------------------------
// Host-side orchestration
// ---------------------------------------------------------------------------
extern "C" void kernel_launch(void* const* d_in, const int* in_sizes, int n_in,
                              void* d_out, int out_size, void* d_ws, size_t ws_size,
                              hipStream_t stream) {
    (void)in_sizes; (void)n_in; (void)out_size; (void)ws_size;

    const float* x     = (const float*)d_in[0];
    const float* h0    = (const float*)d_in[1];   // [2L][H] = [4][768]
    const float* w_ih0 = (const float*)d_in[2];
    const float* w_hh0 = (const float*)d_in[3];
    const float* b_ih0 = (const float*)d_in[4];
    const float* b_hh0 = (const float*)d_in[5];
    const float* w_ih1 = (const float*)d_in[6];
    const float* w_hh1 = (const float*)d_in[7];
    const float* b_ih1 = (const float*)d_in[8];
    const float* b_hh1 = (const float*)d_in[9];
    const float* fc_w  = (const float*)d_in[10];
    const float* fc_b  = (const float*)d_in[11];
    float* out = (float*)d_out;

    // Workspace carve-out (~254 MB).
    char*  ws  = (char*)d_ws;
    size_t off = 0;
    auto alloc = [&](size_t bytes) -> char* {
        off = (off + 255) & ~(size_t)255;
        char* p = ws + off;
        off += bytes;
        return p;
    };
    __bf16* Xb    = (__bf16*)alloc((size_t)T_STEPS * DIN * 2);
    __bf16* W0b   = (__bf16*)alloc((size_t)2 * G3 * DIN * 2);
    __bf16* W1b   = (__bf16*)alloc((size_t)2 * G3 * H2 * 2);
    __bf16* H0b   = (__bf16*)alloc((size_t)T_STEPS * H2 * 2);
    float*  GI    = (float*) alloc((size_t)T_STEPS * 2 * G3 * 4);  // layer0, reused by layer1
    float*  Hseq0 = (float*) alloc((size_t)T_STEPS * H2 * 4);
    float*  hlast = (float*) alloc((size_t)H2 * 4);
    float*  hbuf  = (float*) alloc((size_t)2 * H * 4);
    int*    sync  = (int*)   alloc(16 * 4);

    // Phase counters must start at 0 each call (ws is poisoned once).
    k_zero_ints<<<1, 64, 0, stream>>>(sync, 16);

    const int cvtBlocks = 2048;
    k_f32_to_bf16<<<cvtBlocks, 256, 0, stream>>>(x,     Xb,  (size_t)T_STEPS * DIN);
    k_f32_to_bf16<<<cvtBlocks, 256, 0, stream>>>(w_ih0, W0b, (size_t)2 * G3 * DIN);
    k_f32_to_bf16<<<cvtBlocks, 256, 0, stream>>>(w_ih1, W1b, (size_t)2 * G3 * H2);

    // Layer-0 input GEMM: GI0[t][dir][g] = w_ih0[dir] @ x_t + b_ih0[dir].
    dim3 gemmGrid(T_STEPS / 128, G3 / 128);
    for (int dir = 0; dir < 2; ++dir)
        k_wmma_gemm_bias<<<gemmGrid, 256, 0, stream>>>(
            Xb, W0b + (size_t)dir * G3 * DIN, b_ih0 + (size_t)dir * G3,
            GI + (size_t)dir * G3, 2 * G3, T_STEPS, DIN);

    // Layer-0 recurrence over all T (stores h for every t).
    k_gru_recurrence<<<2 * NWG_REC, 256, 0, stream>>>(
        GI, w_hh0, b_hh0, h0 /* rows 0,1 */, Hseq0, hbuf, sync + 0,
        /*store_all=*/1, T_STEPS);

    // Layer-1 input GEMM over the layer-0 output sequence.
    k_f32_to_bf16<<<cvtBlocks, 256, 0, stream>>>(Hseq0, H0b, (size_t)T_STEPS * H2);
    for (int dir = 0; dir < 2; ++dir)
        k_wmma_gemm_bias<<<gemmGrid, 256, 0, stream>>>(
            H0b, W1b + (size_t)dir * G3 * H2, b_ih1 + (size_t)dir * G3,
            GI + (size_t)dir * G3, 2 * G3, T_STEPS, H2);

    // Layer-1 recurrence (only final h needed).
    k_gru_recurrence<<<2 * NWG_REC, 256, 0, stream>>>(
        GI, w_hh1, b_hh1, h0 + 2 * H /* rows 2,3 */, hlast, hbuf, sync + 4,
        /*store_all=*/0, T_STEPS);

    // Output head.
    k_fc_sigmoid<<<1, 256, 0, stream>>>(hlast, fc_w, fc_b, out);
}
